// RowLSTM_63110249447497
// MI455X (gfx1250) — matrix operand order repaired and checked by reading, not compiled, not run on hardware
//
#include <hip/hip_runtime.h>

// ---------------------------------------------------------------------------
// RowLSTM for MI455X (gfx1250).
// All GEMMs on v_wmma_f32_16x16x32_bf16 (bf16 operands, f32 accum, f32 cell).
//  - Weights pre-swizzled into WMMA A-fragment order -> A frag = 2x global b128.
//  - Activations channel-innermost (+zero halo rows) -> B frag = 2x global b128,
//    no LDS staging, no barriers, no guards in GEMM loops.
//  - K loops split into tap-constant segments (all addresses become immediate
//    offsets) and software-pipelined depth 2 (loads run ahead of WMMA waits).
// ---------------------------------------------------------------------------

#define FEATN 192
#define BB    16
#define HH    64
#define WW    64
#define WP    66      // padded width (halo rows 0 and 65 are zeros)
#define CINN  384
#define CGATE 768

typedef __attribute__((ext_vector_type(16))) __bf16 v16bf;
typedef __attribute__((ext_vector_type(8)))  float  v8f;

__device__ inline unsigned short f32_to_bf16(float f) {
    unsigned u = __builtin_bit_cast(unsigned, f);
    u += 0x7FFFu + ((u >> 16) & 1u);      // round-to-nearest-even
    return (unsigned short)(u >> 16);
}
__device__ inline float bf16_to_f32(unsigned short h) {
    unsigned u = ((unsigned)h) << 16;
    return __builtin_bit_cast(float, u);
}

__device__ inline v16bf ldfrag(const unsigned short* p) {   // 32B = 2x b128
    union { v16bf v; uint4 q[2]; } f;
    f.q[0] = *(const uint4*)p;
    f.q[1] = *(const uint4*)(p + 8);
    return f.v;
}
__device__ inline v8f wmma_bf16(v16bf a, v16bf b, v8f c) {
    return __builtin_amdgcn_wmma_f32_16x16x32_bf16(false, a, false, b, (short)0,
                                                   c, false, false);
}

// Packed A-fragment index for weight matrix Wp[M][K] (M%16==0, K%32==0).
// Per 16x32 block: lane = (m&15) + 16*(k>>4); elem = k&15;
// each lane's 16 bf16 contiguous (32B) -> fragment = 2x b128.
__device__ inline size_t pidx(int m, int k, int K) {
    int mt = m >> 4, rr = m & 15;
    int kc = k >> 5, kl = k & 31;
    int half = kl >> 4;
    int e = kl & 15;
    int lane = rr + (half << 4);
    return ((size_t)(mt * (K >> 5) + kc)) * 512 + (size_t)lane * 16 + (size_t)e;
}

// ---------------------------------------------------------------------------
// k_prep: bake masks + channel shuffle into packed bf16 weights; zero the
// xT halo rows, the hidden-state ping-pong buffers, and the cell state.
// ---------------------------------------------------------------------------
__global__ void k_prep(const float* __restrict__ w_i2s,
                       const float* __restrict__ w_s2s,
                       const float* __restrict__ w_skip,
                       unsigned short* __restrict__ wi2s_p,
                       unsigned short* __restrict__ ws2s_p,
                       unsigned short* __restrict__ wskip_p,
                       unsigned short* __restrict__ xT,
                       unsigned short* __restrict__ hbufT,
                       float* __restrict__ cstT) {
    const int N1 = CGATE * 768;              // i2s weights (packed)
    const int N2 = CGATE * 576;              // s2s weights (packed)
    const int N3 = CINN * FEATN;             // skip weights (packed)
    const int N4 = BB * HH * 2 * CINN;       // xT halo rows (x=0 and x=65)
    const int N5 = 2 * BB * WP * FEATN;      // hidden ping-pong (all zero)
    const int N6 = BB * WW * FEATN;          // cell state
    const int NT = N1 + N2 + N3 + N4 + N5 + N6;
    for (int idx = blockIdx.x * blockDim.x + threadIdx.x; idx < NT;
         idx += gridDim.x * blockDim.x) {
        if (idx < N1) {
            int cp = idx / 768, kt = idx - cp * 768;
            int gate = cp / 192, rem = cp - gate * 192;
            int g = rem / 64, j = rem - g * 64;
            int oc = g * 256 + gate * 64 + j;          // un-shuffled out chan
            int tap = kt / 384, ci = kt - tap * 384;
            float v;
            if (tap == 0) v = w_i2s[((size_t)oc * CINN + ci) * 3 + 0];
            else          v = (g >= ci / 128) ? w_i2s[((size_t)oc * CINN + ci) * 3 + 1] : 0.0f;
            wi2s_p[pidx(cp, kt, 768)] = f32_to_bf16(v);
        } else if (idx < N1 + N2) {
            int t = idx - N1;
            int cg = t / 576, kt = t - cg * 576;
            int tap = kt / 192, cf = kt - tap * 192;
            ws2s_p[pidx(cg, kt, 576)] =
                f32_to_bf16(w_s2s[((size_t)cg * FEATN + cf) * 3 + tap]);
        } else if (idx < N1 + N2 + N3) {
            int t = idx - N1 - N2;
            int c = t / FEATN, cf = t - c * FEATN;
            float v = (c / 128 >= cf / 64) ? w_skip[(size_t)c * FEATN + cf] : 0.0f;
            wskip_p[pidx(c, cf, FEATN)] = f32_to_bf16(v);
        } else if (idx < N1 + N2 + N3 + N4) {
            int t = idx - N1 - N2 - N3;
            int by = t / (2 * CINN);                   // b*HH + y
            int rem = t - by * (2 * CINN);
            int p = rem / CINN, ci = rem - p * CINN;
            int row = p ? (WP - 1) : 0;
            xT[((size_t)by * WP + row) * CINN + ci] = 0;
        } else if (idx < N1 + N2 + N3 + N4 + N5) {
            hbufT[idx - N1 - N2 - N3 - N4] = 0;
        } else {
            cstT[idx - N1 - N2 - N3 - N4 - N5] = 0.0f;
        }
    }
}

// ---------------------------------------------------------------------------
// k_xcastT: transpose-convert x[b][ci][y][x] f32 -> xT[b][y][x+1][ci] bf16
// via a 32x32 LDS tile. Grid (CINN/32, WW/32, B*H), 256 threads.
// ---------------------------------------------------------------------------
__global__ __launch_bounds__(256) void k_xcastT(const float* __restrict__ x,
                                                unsigned short* __restrict__ xT) {
    const int ci0 = blockIdx.x * 32, x0 = blockIdx.y * 32;
    const int by = blockIdx.z;                 // b*HH + y
    const int b = by >> 6, y = by & 63;
    __shared__ unsigned short T[32][33];
    const int t = threadIdx.x;
    {
        int i = t >> 3, j0 = (t & 7) * 4;
        float4 v = *(const float4*)(x + (((size_t)b * CINN + ci0 + i) * HH + y) * WW + x0 + j0);
        T[i][j0 + 0] = f32_to_bf16(v.x);
        T[i][j0 + 1] = f32_to_bf16(v.y);
        T[i][j0 + 2] = f32_to_bf16(v.z);
        T[i][j0 + 3] = f32_to_bf16(v.w);
    }
    __syncthreads();
    {
        int j = t >> 3, i0 = (t & 7) * 4;
        ushort4 o;
        o.x = T[i0 + 0][j]; o.y = T[i0 + 1][j];
        o.z = T[i0 + 2][j]; o.w = T[i0 + 3][j];
        *(ushort4*)(xT + ((size_t)by * WP + x0 + j + 1) * CINN + ci0 + i0) = o;
    }
}

// ---------------------------------------------------------------------------
// GEMM conventions (block = 256 threads = 8 wave32):
//   wave w owns M-subtile (w>>1) and N-subtiles {(w&1)*2, (w&1)*2+1}.
// Inner loops: tap-constant segments, fully unrolled, depth-2 pipelined.
// ---------------------------------------------------------------------------

// k_i2s: h[b, 0:768, y, 0:64] = Wp(768x768) * patches.  No LDS, no barriers.
__global__ __launch_bounds__(256) void k_i2s(const unsigned short* __restrict__ xT,
                                             const unsigned short* __restrict__ wpk,
                                             unsigned short* __restrict__ h) {
    const int cb = blockIdx.x, y = blockIdx.y, b = blockIdx.z;
    const int tid = threadIdx.x, wv = tid >> 5, lane = tid & 31;
    const int mt = wv >> 1, ntBase = (wv & 1) * 2;
    const int half = lane >> 4, n = lane & 15;
    const int col = ntBase * 16 + n;

    const unsigned short* abase = wpk + (size_t)(cb * 4 + mt) * 24 * 512 + lane * 16;
    const unsigned short* xbase = xT + ((size_t)b * HH + y) * WP * CINN;
    v8f acc0 = {}, acc1 = {};

#pragma unroll
    for (int tap = 0; tap < 2; ++tap) {        // causal taps: x-1, x
        const unsigned short* ap = abase + (size_t)tap * (384 * 16);
        const unsigned short* bp = xbase + (size_t)(col + tap) * CINN + 16 * half;
        v16bf fa = ldfrag(ap);
        v16bf f0 = ldfrag(bp);
        v16bf f1 = ldfrag(bp + 16 * CINN);
#pragma unroll
        for (int kk = 0; kk < 12; ++kk) {
            v16bf na = {}, n0 = {}, n1 = {};
            if (kk + 1 < 12) {
                na = ldfrag(ap + (kk + 1) * 512);
                n0 = ldfrag(bp + (kk + 1) * 32);
                n1 = ldfrag(bp + (kk + 1) * 32 + 16 * CINN);
            }
            acc0 = wmma_bf16(fa, f0, acc0);
            acc1 = wmma_bf16(fa, f1, acc1);
            if (kk + 1 < 12) { fa = na; f0 = n0; f1 = n1; }
        }
    }
#pragma unroll
    for (int v = 0; v < 8; ++v) {
        int m = mt * 16 + v + 8 * half;
        size_t base = (((size_t)b * CGATE + (cb * 64 + m)) * HH + y) * WW;
        h[base + ntBase * 16 + n]       = f32_to_bf16(acc0[v]);
        h[base + (ntBase + 1) * 16 + n] = f32_to_bf16(acc1[v]);
    }
}

// ---------------------------------------------------------------------------
// k_row: one scan row r. Grid (12 j-blocks, 16 b). WG computes s for 16
// features x 4 gates (M=64, mtile = gate*12+jb) x N=64, K=576; accumulator
// seeded from h; fused sigmoid/tanh LSTM epilogue via padded LDS exchange.
// ---------------------------------------------------------------------------
__global__ __launch_bounds__(256) void k_row(const unsigned short* __restrict__ wpk,
                                             const unsigned short* __restrict__ h,
                                             const unsigned short* __restrict__ hprevT,
                                             unsigned short* __restrict__ hcurT,
                                             float* __restrict__ cstT,
                                             unsigned short* __restrict__ outhT,
                                             int r) {
    const int jb = blockIdx.x, b = blockIdx.y;
    __shared__ float Sl[64 * 65];                    // padded: no bank conflicts
    const int tid = threadIdx.x, wv = tid >> 5, lane = tid & 31;
    const int mt = wv >> 1, ntBase = (wv & 1) * 2;   // mt == gate
    const int half = lane >> 4, n = lane & 15;
    const int col = ntBase * 16 + n;

    const unsigned short* abase = wpk + (size_t)(mt * 12 + jb) * 18 * 512 + lane * 16;
    const unsigned short* hbase = hprevT + (size_t)b * WP * FEATN;

    v8f acc0, acc1;                                  // C init = input term h
#pragma unroll
    for (int v = 0; v < 8; ++v) {
        int m = mt * 16 + v + 8 * half;
        int cg = (m >> 4) * FEATN + jb * 16 + (m & 15);
        size_t base = (((size_t)b * CGATE + cg) * HH + r) * WW;
        acc0[v] = bf16_to_f32(h[base + ntBase * 16 + n]);
        acc1[v] = bf16_to_f32(h[base + (ntBase + 1) * 16 + n]);
    }
#pragma unroll
    for (int tap = 0; tap < 3; ++tap) {              // taps: x-1, x, x+1
        const unsigned short* ap = abase + (size_t)tap * (192 * 16);
        const unsigned short* bp = hbase + (size_t)(col + tap) * FEATN + 16 * half;
        v16bf fa = ldfrag(ap);
        v16bf f0 = ldfrag(bp);
        v16bf f1 = ldfrag(bp + 16 * FEATN);
#pragma unroll
        for (int kk = 0; kk < 6; ++kk) {
            v16bf na = {}, n0 = {}, n1 = {};
            if (kk + 1 < 6) {
                na = ldfrag(ap + (kk + 1) * 512);
                n0 = ldfrag(bp + (kk + 1) * 32);
                n1 = ldfrag(bp + (kk + 1) * 32 + 16 * FEATN);
            }
            acc0 = wmma_bf16(fa, f0, acc0);
            acc1 = wmma_bf16(fa, f1, acc1);
            if (kk + 1 < 6) { fa = na; f0 = n0; f1 = n1; }
        }
    }
#pragma unroll
    for (int v = 0; v < 8; ++v) {
        int m = mt * 16 + v + 8 * half;
        Sl[m * 65 + ntBase * 16 + n]       = acc0[v];
        Sl[m * 65 + (ntBase + 1) * 16 + n] = acc1[v];
    }
    __syncthreads();
    // fused LSTM: thread = one x column, 4 consecutive features (vector IO)
    {
        const int xx = tid >> 2, jj0 = (tid & 3) * 4;
        const int jgl0 = jb * 16 + jj0;
        size_t cbase = ((size_t)b * WW + xx) * FEATN + jgl0;
        float4 cold = *(const float4*)(cstT + cbase);
        float co[4] = {cold.x, cold.y, cold.z, cold.w};
        float4 cnew;
        ushort4 hb;
        float cn[4], hn[4];
#pragma unroll
        for (int i = 0; i < 4; ++i) {
            int jj = jj0 + i;
            float so = Sl[(0 * 16 + jj) * 65 + xx];
            float sf = Sl[(1 * 16 + jj) * 65 + xx];
            float si = Sl[(2 * 16 + jj) * 65 + xx];
            float sg = Sl[(3 * 16 + jj) * 65 + xx];
            float o  = 1.0f / (1.0f + expf(-so));
            float f  = 1.0f / (1.0f + expf(-sf));
            float ii = 1.0f / (1.0f + expf(-si));
            float g  = 1.0f / (1.0f + expf(-sg));    // reference sigmoids g too
            cn[i] = f * co[i] + ii * g;
            hn[i] = o * tanhf(cn[i]);
        }
        cnew.x = cn[0]; cnew.y = cn[1]; cnew.z = cn[2]; cnew.w = cn[3];
        hb.x = f32_to_bf16(hn[0]); hb.y = f32_to_bf16(hn[1]);
        hb.z = f32_to_bf16(hn[2]); hb.w = f32_to_bf16(hn[3]);
        *(float4*)(cstT + cbase) = cnew;
        *(ushort4*)(hcurT + ((size_t)b * WP + xx + 1) * FEATN + jgl0) = hb;
        *(ushort4*)(outhT + (((size_t)b * HH + r) * WW + xx) * FEATN + jgl0) = hb;
    }
}

// ---------------------------------------------------------------------------
// k_skip: masked 1x1 conv + bias + residual. Grid (CIN/64, H, B), K=192.
// ---------------------------------------------------------------------------
__global__ __launch_bounds__(256) void k_skip(const float* __restrict__ x,
                                              const unsigned short* __restrict__ wpk,
                                              const float* __restrict__ bskip,
                                              const unsigned short* __restrict__ outhT,
                                              float* __restrict__ out) {
    const int cb = blockIdx.x, y = blockIdx.y, b = blockIdx.z;
    const int tid = threadIdx.x, wv = tid >> 5, lane = tid & 31;
    const int mt = wv >> 1, ntBase = (wv & 1) * 2;
    const int half = lane >> 4, n = lane & 15;
    const int col = ntBase * 16 + n;

    const unsigned short* ap = wpk + (size_t)(cb * 4 + mt) * 6 * 512 + lane * 16;
    const unsigned short* bp = outhT + (((size_t)b * HH + y) * WW + col) * FEATN + 16 * half;
    v8f acc0 = {}, acc1 = {};

    v16bf fa = ldfrag(ap);
    v16bf f0 = ldfrag(bp);
    v16bf f1 = ldfrag(bp + 16 * FEATN);
#pragma unroll
    for (int kk = 0; kk < 6; ++kk) {
        v16bf na = {}, n0 = {}, n1 = {};
        if (kk + 1 < 6) {
            na = ldfrag(ap + (kk + 1) * 512);
            n0 = ldfrag(bp + (kk + 1) * 32);
            n1 = ldfrag(bp + (kk + 1) * 32 + 16 * FEATN);
        }
        acc0 = wmma_bf16(fa, f0, acc0);
        acc1 = wmma_bf16(fa, f1, acc1);
        if (kk + 1 < 6) { fa = na; f0 = n0; f1 = n1; }
    }
#pragma unroll
    for (int v = 0; v < 8; ++v) {
        int m = mt * 16 + v + 8 * half;
        int c = cb * 64 + m;
        float bias = bskip[c];
        size_t base = (((size_t)b * CINN + c) * HH + y) * WW;
        int x0 = ntBase * 16 + n, x1 = (ntBase + 1) * 16 + n;
        out[base + x0] = acc0[v] + x[base + x0] + bias;
        out[base + x1] = acc1[v] + x[base + x1] + bias;
    }
}

// ---------------------------------------------------------------------------
extern "C" void kernel_launch(void* const* d_in, const int* in_sizes, int n_in,
                              void* d_out, int out_size, void* d_ws, size_t ws_size,
                              hipStream_t stream) {
    const float* x      = (const float*)d_in[0];
    const float* w_i2s  = (const float*)d_in[1];
    const float* w_s2s  = (const float*)d_in[2];
    const float* w_skip = (const float*)d_in[3];
    const float* b_skip = (const float*)d_in[4];
    float* out = (float*)d_out;
    (void)in_sizes; (void)n_in; (void)out_size;

    char* ws = (char*)d_ws;
    size_t off = 0;
    auto take = [&](size_t bytes) -> char* {
        char* p = ws + off;
        off = (off + bytes + 255) & ~(size_t)255;
        return p;
    };
    unsigned short* h     = (unsigned short*)take((size_t)BB * CGATE * HH * WW * 2); // shuffled conv out (bf16)
    unsigned short* xT    = (unsigned short*)take((size_t)BB * HH * WP * CINN * 2);  // x, chan-innermost + halo
    unsigned short* outhT = (unsigned short*)take((size_t)BB * HH * WW * FEATN * 2); // scan out, chan-innermost
    unsigned short* hbufT = (unsigned short*)take((size_t)2 * BB * WP * FEATN * 2);  // hidden ping-pong + halo
    float*          cstT  = (float*)take((size_t)BB * WW * FEATN * 4);               // cell state (f32)
    unsigned short* wi2p  = (unsigned short*)take((size_t)CGATE * 768 * 2);
    unsigned short* ws2p  = (unsigned short*)take((size_t)CGATE * 576 * 2);
    unsigned short* wskp  = (unsigned short*)take((size_t)CINN * FEATN * 2);
    if (off > ws_size) return;  // defensive: workspace too small

    k_prep<<<dim3(2048), 256, 0, stream>>>(w_i2s, w_s2s, w_skip, wi2p, ws2p, wskp,
                                           xT, hbufT, cstT);
    k_xcastT<<<dim3(CINN / 32, WW / 32, BB * HH), 256, 0, stream>>>(x, xT);
    k_i2s<<<dim3(CGATE / 64, HH, BB), 256, 0, stream>>>(xT, wi2p, h);

    const size_t HS = (size_t)BB * WP * FEATN;
    for (int r = 0; r < HH; ++r) {
        const unsigned short* hrd = hbufT + ((r & 1) ? HS : 0);
        unsigned short*       hwr = hbufT + ((r & 1) ? 0 : HS);
        k_row<<<dim3(FEATN / 16, BB), 256, 0, stream>>>(ws2p, h, hrd, hwr, cstT, outhT, r);
    }
    k_skip<<<dim3(CINN / 64, HH, BB), 256, 0, stream>>>(x, wskp, b_skip, outhT, out);
}
